// SelfAttentionBlock_86225763434949
// MI455X (gfx1250) — compile-verified, matrix operand order checked by reading, not run on hardware
//
#include <hip/hip_runtime.h>
#include <hip/hip_bf16.h>

typedef __attribute__((ext_vector_type(16))) _Float16 v16h;
typedef __attribute__((ext_vector_type(8)))  _Float16 v8h;
typedef __attribute__((ext_vector_type(8)))  float    v8f;

#define N_TOK 4096
#define C_IN  512
#define CH    256
#define C_OUT 512
#define NB    4

// ---- WMMA fragment loader (CDNA5 16-bit 16x32 layout, wave32) ----
// storage is (row, k)-major: elem(r,k) at base[r*stride + k].
// Per lane: two contiguous 8-half (16B) runs -> compiler emits ds_load_b128.
__device__ __forceinline__ v16h frag_rk(const _Float16* base, int stride) {
  const int lane = threadIdx.x & 31;
  const int r = lane & 15;
  const int h = lane >> 4;
  v16h f;
#pragma unroll
  for (int j = 0; j < 8; ++j) {
    const int k = ((j >> 2) << 4) + (h << 3) + ((j & 3) << 1);
    f[2*j]   = base[r*stride + k];
    f[2*j+1] = base[r*stride + k + 1];
  }
  return f;
}

__device__ __forceinline__ v8f wmma16(v16h a, v16h b, v8f c) {
  return __builtin_amdgcn_wmma_f32_16x16x32_f16(false, a, false, b, (short)0, c,
                                                false, false);
}

// ===================== 1x1-conv projection: [C_IN,N] x W^T -> f16 [N,CH] ====
#define XT_STRIDE 40   // Xs stored transposed: [token][channel], 80B rows
#define WS_STRIDE 40

__global__ __launch_bounds__(256) void proj_kernel(
    const float* __restrict__ X, const float* __restrict__ W,
    const float* __restrict__ bias, _Float16* __restrict__ out) {
  __shared__ _Float16 Xs[64 * XT_STRIDE];
  __shared__ _Float16 Ws[CH * WS_STRIDE];
  const int tid   = threadIdx.x;
  const int b     = blockIdx.x >> 6;
  const int nbase = (blockIdx.x & 63) << 6;   // 64 tokens per WG
  const int w     = tid >> 5;
  const int roff  = (w & 3) << 4;             // wave row tile
  const int coff  = (w >> 2) << 7;            // wave col tile (0 / 128)
  const int lane  = tid & 31;
  const int h     = lane >> 4;
  const int nloc  = lane & 15;

  v8f acc[8] = {};

  const float* Xb = X + (size_t)b * C_IN * N_TOK;
  for (int kc = 0; kc < C_IN; kc += 32) {
    __syncthreads();
    // input tile: read [channel][token] coalesced, store transposed [token][channel]
    for (int idx = tid; idx < 32 * 64; idx += 256) {
      int c = idx >> 6, n = idx & 63;
      Xs[n * XT_STRIDE + c] = (_Float16)Xb[(size_t)(kc + c) * N_TOK + nbase + n];
    }
    // weight tile: 16B vector copies (row stride 80B is 16B-aligned)
    for (int idx = tid; idx < CH * 4; idx += 256) {
      int o = idx >> 2, c8 = (idx & 3) << 3;
      const float* src = W + o * C_IN + kc + c8;
      v8h tmp;
#pragma unroll
      for (int e = 0; e < 8; ++e) tmp[e] = (_Float16)src[e];
      *(v8h*)&Ws[o * WS_STRIDE + c8] = tmp;
    }
    __syncthreads();
    v16h a = frag_rk(Xs + roff * XT_STRIDE, XT_STRIDE);   // A(m=token, k=channel)
#pragma unroll
    for (int t = 0; t < 8; ++t) {
      v16h bf = frag_rk(Ws + (coff + t * 16) * WS_STRIDE, WS_STRIDE);
      acc[t] = wmma16(a, bf, acc[t]);
    }
  }
#pragma unroll
  for (int t = 0; t < 8; ++t) {
    int o = coff + t * 16 + nloc;
    float bval = bias[o];
#pragma unroll
    for (int i = 0; i < 8; ++i) {
      int q = nbase + roff + h * 8 + i;
      out[((size_t)b * N_TOK + q) * CH + o] = (_Float16)(acc[t][i] + bval);
    }
  }
}

// ===================== flash attention: Q[N,CH] K[N,CH] V[N,CH] -> ctx ======
#define KS_STRIDE 272  // [key][channel], 544B rows (16B multiple)
#define VT_STRIDE 40   // V transposed: [channel][key], 80B rows
#define PS_STRIDE 40

__global__ __launch_bounds__(128) void flash_kernel(
    const _Float16* __restrict__ Qh, const _Float16* __restrict__ Kh,
    const _Float16* __restrict__ Vh, const unsigned char* __restrict__ mask,
    _Float16* __restrict__ ctx) {
  __shared__ _Float16 KS[32 * KS_STRIDE];        // K tile  [32 keys][256 ch]
  __shared__ _Float16 VT[CH * VT_STRIDE];        // V tile  [256 ch][32 keys]
  __shared__ _Float16 PS[4 * 16 * PS_STRIDE];
  __shared__ float    MS[32];
  const int tid   = threadIdx.x;
  const int b     = blockIdx.x >> 6;
  const int qbase = (blockIdx.x & 63) << 6;
  const int w     = tid >> 5;
  const int lane  = tid & 31;
  const int h     = lane >> 4;
  const int nloc  = lane & 15;
  const int q0    = qbase + w * 16;              // 16 queries per wave

  // Q fragments (16 x 256) held in registers for the whole kernel
  v16h qa[8];
  const _Float16* Qrow = Qh + ((size_t)b * N_TOK + q0) * CH;
#pragma unroll
  for (int kc = 0; kc < 8; ++kc) qa[kc] = frag_rk(Qrow + kc * 32, CH);

  v8f oacc[16] = {};
  float mrow[8], lrow[8];
#pragma unroll
  for (int i = 0; i < 8; ++i) { mrow[i] = -1e30f; lrow[i] = 0.0f; }

  _Float16* psw = PS + w * 16 * PS_STRIDE;
  const float scale = 0.0625f;                   // CH^-0.5 = 1/16

  for (int kt = 0; kt < N_TOK; kt += 32) {
    __syncthreads();
    // K tile: 16B vector copies, layout [key][channel]
    for (int idx = tid; idx < 32 * (CH / 8); idx += 128) {
      int kk = idx >> 5, c8 = (idx & 31) << 3;
      v8h kv = *(const v8h*)(Kh + ((size_t)b * N_TOK + kt + kk) * CH + c8);
      *(v8h*)&KS[kk * KS_STRIDE + c8] = kv;
    }
    // V tile transposed: [channel][key] (scalar scatter; coalesced global reads)
    for (int idx = tid; idx < 32 * CH; idx += 128) {
      int kk = idx >> 8, cc = idx & 255;
      VT[cc * VT_STRIDE + kk] = Vh[((size_t)b * N_TOK + kt + kk) * CH + cc];
    }
    if (tid < 32) MS[tid] = mask[(size_t)b * N_TOK + kt + tid] ? 1.0f : 0.0f;
    __syncthreads();

    // S = Q K^T for 32 keys (two 16x16 f32 tiles); B(k=channel, n=key)
    v8f c0 = {}, c1 = {};
#pragma unroll
    for (int kc = 0; kc < 8; ++kc) {
      c0 = wmma16(qa[kc], frag_rk(KS + kc * 32, KS_STRIDE), c0);
      c1 = wmma16(qa[kc], frag_rk(KS + 16 * KS_STRIDE + kc * 32, KS_STRIDE), c1);
    }

    const float mv0 = MS[nloc];
    const float mv1 = MS[16 + nloc];
    float alpha[8];
#pragma unroll
    for (int i = 0; i < 8; ++i) {
      float s0 = (mv0 != 0.0f) ? c0[i] * scale : -1e7f;
      float s1 = (mv1 != 0.0f) ? c1[i] * scale : -1e7f;
      float r = fmaxf(s0, s1);
#pragma unroll
      for (int d = 1; d < 16; d <<= 1) r = fmaxf(r, __shfl_xor(r, d, 32));
      float mnew = fmaxf(mrow[i], r);
      alpha[i] = __expf(mrow[i] - mnew);
      float p0 = __expf(s0 - mnew);
      float p1 = __expf(s1 - mnew);
      float rs = p0 + p1;
#pragma unroll
      for (int d = 1; d < 16; d <<= 1) rs += __shfl_xor(rs, d, 32);
      lrow[i] = lrow[i] * alpha[i] + rs;
      mrow[i] = mnew;
      // bounce P through LDS (C-layout -> A-layout transpose)
      psw[(h * 8 + i) * PS_STRIDE + nloc]      = (_Float16)p0;
      psw[(h * 8 + i) * PS_STRIDE + 16 + nloc] = (_Float16)p1;
    }
#pragma unroll
    for (int t = 0; t < 16; ++t)
#pragma unroll
      for (int i = 0; i < 8; ++i) oacc[t][i] *= alpha[i];

    // per-wave LDS RAW: make P stores visible before A-frag reads
    asm volatile("s_wait_dscnt 0" ::: "memory");
    v16h pa = frag_rk(psw, PS_STRIDE);           // A(m=query, k=key)
#pragma unroll
    for (int t = 0; t < 16; ++t) {
      // B(k=key, n=channel) from transposed V: row-contiguous -> ds_load_b128
      v16h vb = frag_rk(VT + (t * 16) * VT_STRIDE, VT_STRIDE);
      oacc[t] = wmma16(pa, vb, oacc[t]);
    }
  }

#pragma unroll
  for (int i = 0; i < 8; ++i) {
    float inv = 1.0f / lrow[i];
    size_t row = ((size_t)b * N_TOK + q0 + h * 8 + i) * CH;
#pragma unroll
    for (int t = 0; t < 16; ++t)
      ctx[row + t * 16 + nloc] = (_Float16)(oacc[t][i] * inv);
  }
}

// ===================== output projection: ctx[N,CH] x Wo^T -> out[C_OUT,N] ==
#define CS_STRIDE  40
#define WOS_STRIDE 40

__global__ __launch_bounds__(256) void outproj_kernel(
    const _Float16* __restrict__ ctx, const float* __restrict__ Wo,
    const float* __restrict__ bo, float* __restrict__ out) {
  __shared__ _Float16 Cs[64 * CS_STRIDE];
  __shared__ _Float16 Wos[256 * WOS_STRIDE];
  const int tid   = threadIdx.x;
  const int b     = blockIdx.x >> 7;
  const int rem   = blockIdx.x & 127;
  const int qbase = (rem >> 1) << 6;
  const int obase = (rem & 1) << 8;
  const int w     = tid >> 5;
  const int roff  = (w & 3) << 4;
  const int coff  = (w >> 2) << 7;
  const int lane  = tid & 31;
  const int h     = lane >> 4;
  const int nloc  = lane & 15;

  v8f acc[8] = {};
  for (int kc = 0; kc < CH; kc += 32) {
    __syncthreads();
    // ctx tile [64 q][32 c]: 16B vector copies
    for (int idx = tid; idx < 64 * 4; idx += 256) {
      int q = idx >> 2, c8 = (idx & 3) << 3;
      v8h t8 = *(const v8h*)(ctx + ((size_t)b * N_TOK + qbase + q) * CH + kc + c8);
      *(v8h*)&Cs[q * CS_STRIDE + c8] = t8;
    }
    // Wo tile [256 o][32 c]: f32->f16 convert, 16B vector stores
    for (int idx = tid; idx < 256 * 4; idx += 256) {
      int o = idx >> 2, c8 = (idx & 3) << 3;
      const float* src = Wo + (obase + o) * CH + kc + c8;
      v8h tmp;
#pragma unroll
      for (int e = 0; e < 8; ++e) tmp[e] = (_Float16)src[e];
      *(v8h*)&Wos[o * WOS_STRIDE + c8] = tmp;
    }
    __syncthreads();
    v16h a = frag_rk(Cs + roff * CS_STRIDE, CS_STRIDE);
#pragma unroll
    for (int t = 0; t < 8; ++t) {
      v16h bf = frag_rk(Wos + (coff + t * 16) * WOS_STRIDE, WOS_STRIDE);
      acc[t] = wmma16(a, bf, acc[t]);
    }
  }
#pragma unroll
  for (int t = 0; t < 8; ++t) {
    int o = obase + coff + t * 16 + nloc;
    float bval = bo[o];
#pragma unroll
    for (int i = 0; i < 8; ++i) {
      int q = qbase + roff + h * 8 + i;
      out[((size_t)b * C_OUT + o) * N_TOK + q] = acc[t][i] + bval;
    }
  }
}

// ============================================================================
extern "C" void kernel_launch(void* const* d_in, const int* in_sizes, int n_in,
                              void* d_out, int out_size, void* d_ws, size_t ws_size,
                              hipStream_t stream) {
  const float* qf = (const float*)d_in[0];
  const float* kf = (const float*)d_in[1];
  const float* vf = (const float*)d_in[2];
  const unsigned char* mask = (const unsigned char*)d_in[3];
  const float* Wq = (const float*)d_in[4];
  const float* bq = (const float*)d_in[5];
  const float* Wk = (const float*)d_in[6];
  const float* bk = (const float*)d_in[7];
  const float* Wv = (const float*)d_in[8];
  const float* bv = (const float*)d_in[9];
  const float* Wo = (const float*)d_in[10];
  const float* bo = (const float*)d_in[11];
  float* out = (float*)d_out;

  _Float16* ws = (_Float16*)d_ws;
  const size_t per = (size_t)NB * N_TOK * CH;   // 4,194,304 halfs each
  _Float16* Qh = ws;
  _Float16* Kh = ws + per;
  _Float16* Vh = ws + 2 * per;
  _Float16* CT = ws + 3 * per;

  dim3 pg(NB * (N_TOK / 64));                   // 256 WGs
  proj_kernel<<<pg, 256, 0, stream>>>(qf, Wq, bq, Qh);
  proj_kernel<<<pg, 256, 0, stream>>>(kf, Wk, bk, Kh);
  proj_kernel<<<pg, 256, 0, stream>>>(vf, Wv, bv, Vh);
  flash_kernel<<<pg, 128, 0, stream>>>(Qh, Kh, Vh, mask, CT);
  outproj_kernel<<<dim3(NB * (N_TOK / 64) * 2), 256, 0, stream>>>(CT, Wo, bo, out);
}
